// FusionFeature_Layer_61435212201999
// MI455X (gfx1250) — matrix operand order repaired and checked by reading, not compiled, not run on hardware
//
#include <hip/hip_runtime.h>

typedef __attribute__((ext_vector_type(16))) _Float16 v16h;
typedef __attribute__((ext_vector_type(8)))  float    v8f;

#define BATCH 8
#define CDIM  128
#define NDIM  3136          // 56*56
#define BM    64
#define BN    64
#define NBLK  (NDIM / BM)   // 49

union H16 { uint4 u[2]; v16h v; };

// 16 contiguous halves -> B-fragment (element e -> K = 16*kg + e)
__device__ __forceinline__ v16h ld_bfrag(const _Float16* p) {
  H16 t;
  t.u[0] = *(const uint4*)(p);
  t.u[1] = *(const uint4*)(p + 8);
  return t.v;
}
// A-fragment: 16 contiguous halves at p (K = kb+16kg+{0..15})
__device__ __forceinline__ v16h ld_afrag(const _Float16* p) {
  H16 t;
  t.u[0] = *(const uint4*)(p);
  t.u[1] = *(const uint4*)(p + 8);
  return t.v;
}

__device__ __forceinline__ float redmax16(float v) {
  v = fmaxf(v, __shfl_xor(v, 1, 16));
  v = fmaxf(v, __shfl_xor(v, 2, 16));
  v = fmaxf(v, __shfl_xor(v, 4, 16));
  v = fmaxf(v, __shfl_xor(v, 8, 16));
  return v;
}
__device__ __forceinline__ float redsum16(float v) {
  v += __shfl_xor(v, 1, 16);
  v += __shfl_xor(v, 2, 16);
  v += __shfl_xor(v, 4, 16);
  v += __shfl_xor(v, 8, 16);
  return v;
}

__global__ __launch_bounds__(128)
void fused_feature_attn(const float* __restrict__ qg,   // query  [B, C*N] flat
                        const float* __restrict__ sg,   // support[B, C*N] flat
                        float* __restrict__ outq,
                        float* __restrict__ outs) {
  const int blk  = blockIdx.x;
  const int b    = blk / NBLK;
  const int m0   = (blk % NBLK) * BM;
  const int tid  = threadIdx.x;
  const int lane = tid & 31;
  const int wave = tid >> 5;      // 0..3
  const int kg   = lane >> 4;     // half-wave group (0/1)
  const int ln   = lane & 15;

  const float* qb = qg + (size_t)b * CDIM * NDIM;
  const float* sb = sg + (size_t)b * CDIM * NDIM;

  __shared__ __align__(16) _Float16 Qs[BM * CDIM];      // [row][c]
  __shared__ __align__(16) _Float16 Ks[BN * CDIM];      // [n_local][c]   (sf tile transposed)
  __shared__ __align__(16) _Float16 Vs[CDIM * BN];      // [c][n_local]   (qf tile transposed)
  __shared__ __align__(16) _Float16 Ps[4][16 * BN];     // per-wave P tile [row][col]

  // ---- load Q row-block (fp32 -> f16) into LDS ----
  #pragma unroll
  for (int e = tid * 4; e < BM * CDIM; e += 128 * 4) {
    const int row = e / CDIM, c = e % CDIM;
    const float4 f = *(const float4*)(qb + (size_t)(m0 + row) * CDIM + c);
    Qs[e + 0] = (_Float16)f.x; Qs[e + 1] = (_Float16)f.y;
    Qs[e + 2] = (_Float16)f.z; Qs[e + 3] = (_Float16)f.w;
  }
  __syncthreads();

  // ---- per-wave Q A-fragments (16 rows x 128 K, 4 k-steps of 32) ----
  v16h qa[4];
  const _Float16* qrow = Qs + (size_t)(wave * 16 + ln) * CDIM;
  #pragma unroll
  for (int k = 0; k < 4; ++k)
    qa[k] = ld_afrag(qrow + k * 32 + kg * 16);

  // ---- flash-attention state ----
  v8f o[8];
  #pragma unroll
  for (int jc = 0; jc < 8; ++jc) o[jc] = (v8f){0.f,0.f,0.f,0.f,0.f,0.f,0.f,0.f};
  float mrow[8], lrow[8], rsum[8];
  #pragma unroll
  for (int r = 0; r < 8; ++r) { mrow[r] = -INFINITY; lrow[r] = 0.f; rsum[r] = 0.f; }

  for (int n0 = 0; n0 < NDIM; n0 += BN) {
    __syncthreads();   // previous iter's K/V readers done before overwrite

    // K tile: sf[c, n0+n] -> Ks[n][c]
    for (int e = tid; e < CDIM * (BN / 4); e += 128) {
      const int c  = e >> 4;
      const int n4 = (e & 15) * 4;
      const float4 f = *(const float4*)(sb + (size_t)c * NDIM + n0 + n4);
      Ks[(n4 + 0) * CDIM + c] = (_Float16)f.x;
      Ks[(n4 + 1) * CDIM + c] = (_Float16)f.y;
      Ks[(n4 + 2) * CDIM + c] = (_Float16)f.z;
      Ks[(n4 + 3) * CDIM + c] = (_Float16)f.w;
    }
    // V tile: qf[n0+n, c] -> Vs[c][n]
    for (int e = tid; e < BN * (CDIM / 4); e += 128) {
      const int n  = e >> 5;
      const int c4 = (e & 31) * 4;
      const float4 f = *(const float4*)(qb + (size_t)(n0 + n) * CDIM + c4);
      Vs[(c4 + 0) * BN + n] = (_Float16)f.x;
      Vs[(c4 + 1) * BN + n] = (_Float16)f.y;
      Vs[(c4 + 2) * BN + n] = (_Float16)f.z;
      Vs[(c4 + 3) * BN + n] = (_Float16)f.w;
    }
    __syncthreads();

    // ---- S = Q @ K : 4 n-tiles of 16, K-depth 128 ----
    v8f sacc[4];
    #pragma unroll
    for (int j = 0; j < 4; ++j) sacc[j] = (v8f){0.f,0.f,0.f,0.f,0.f,0.f,0.f,0.f};
    #pragma unroll
    for (int k = 0; k < 4; ++k) {
      #pragma unroll
      for (int j = 0; j < 4; ++j) {
        const v16h bf = ld_bfrag(Ks + (size_t)(j * 16 + ln) * CDIM + k * 32 + kg * 16);
        sacc[j] = __builtin_amdgcn_wmma_f32_16x16x32_f16(
            false, qa[k], false, bf, (short)0, sacc[j], false, false);
      }
    }

    // ---- online softmax row update + raw row-sum for the sigmoid mask ----
    float p[4][8];
    #pragma unroll
    for (int r = 0; r < 8; ++r) {
      float mx = sacc[0][r], sm = sacc[0][r];
      #pragma unroll
      for (int j = 1; j < 4; ++j) { mx = fmaxf(mx, sacc[j][r]); sm += sacc[j][r]; }
      mx = redmax16(mx);
      rsum[r] += redsum16(sm);
      const float mnew  = fmaxf(mrow[r], mx);
      const float scale = __expf(mrow[r] - mnew);   // 0 on first block
      float psum = 0.f;
      #pragma unroll
      for (int j = 0; j < 4; ++j) { p[j][r] = __expf(sacc[j][r] - mnew); psum += p[j][r]; }
      lrow[r] = lrow[r] * scale + redsum16(psum);
      mrow[r] = mnew;
      #pragma unroll
      for (int jc = 0; jc < 8; ++jc) o[jc][r] *= scale;
    }

    // ---- P (D-layout) -> LDS -> A-layout; same-wave LDS RAW fence ----
    _Float16* pw = Ps[wave];
    #pragma unroll
    for (int j = 0; j < 4; ++j)
      #pragma unroll
      for (int r = 0; r < 8; ++r)
        pw[(r + kg * 8) * BN + j * 16 + ln] = (_Float16)p[j][r];
    asm volatile("s_wait_dscnt 0x0" ::: "memory");

    // ---- O += P @ V : K-depth 64 (2 k-steps), 8 c-tiles ----
    #pragma unroll
    for (int kt = 0; kt < 2; ++kt) {
      const v16h pa = ld_afrag(pw + ln * BN + kt * 32 + kg * 16);
      #pragma unroll
      for (int jc = 0; jc < 8; ++jc) {
        const v16h vb = ld_bfrag(Vs + (size_t)(jc * 16 + ln) * BN + kt * 32 + kg * 16);
        o[jc] = __builtin_amdgcn_wmma_f32_16x16x32_f16(
            false, pa, false, vb, (short)0, o[jc], false, false);
      }
    }
  }

  // ---- epilogue: normalize, mask, residual, both outputs ----
  float* oq = outq + (size_t)b * CDIM * NDIM;
  float* os = outs + (size_t)b * CDIM * NDIM;
  #pragma unroll
  for (int r = 0; r < 8; ++r) {
    const int   row  = m0 + wave * 16 + kg * 8 + r;
    const float inv  = 1.0f / lrow[r];
    const float mask = 1.0f / (1.0f + __expf(-rsum[r]));
    #pragma unroll
    for (int jc = 0; jc < 8; ++jc) {
      const int    c   = jc * 16 + ln;
      const size_t idx = (size_t)row * CDIM + c;
      oq[idx] = qb[idx] + o[jc][r] * inv * mask;
      os[idx] = sb[idx] * (1.0f + mask);
    }
  }
}

extern "C" void kernel_launch(void* const* d_in, const int* in_sizes, int n_in,
                              void* d_out, int out_size, void* d_ws, size_t ws_size,
                              hipStream_t stream) {
  (void)in_sizes; (void)n_in; (void)out_size; (void)d_ws; (void)ws_size;
  const float* q = (const float*)d_in[0];   // query_feature  [8,128,56,56]
  const float* s = (const float*)d_in[1];   // support_feature[8,128,56,56]
  float* outq = (float*)d_out;
  float* outs = (float*)d_out + (size_t)BATCH * CDIM * NDIM;
  fused_feature_attn<<<dim3(BATCH * NBLK), dim3(128), 0, stream>>>(q, s, outq, outs);
}
